// EX_Module_54812372632024
// MI455X (gfx1250) — compile-verified, matrix operand order checked by reading, not run on hardware
//
#include <hip/hip_runtime.h>
#include <hip/hip_bf16.h>

// ---------------------------------------------------------------------------
// Problem constants (from reference): b=32, c=64, c2=32, h=w=128
// ---------------------------------------------------------------------------
#define B   32
#define C   64
#define C2  32
#define HW  16384
#define CH1 32      // pass1 chunks per batch (512 px each)
#define PX1 512
#define CH2 32      // pass2 chunks per batch
#define PX2 512
#define NK3 16      // pass3 K-chunks per batch (1024 px each)
#define PX3 1024

typedef __attribute__((ext_vector_type(16))) _Float16 v16h;
typedef __attribute__((ext_vector_type(8)))  float    v8f;

// workspace layout (float offsets)
constexpr size_t OFF_Q       = 0;                         // B*HW
constexpr size_t OFF_SP      = OFF_Q + (size_t)B*HW;      // B*HW
constexpr size_t OFF_QMAXP   = OFF_SP + (size_t)B*HW;     // B*CH1
constexpr size_t OFF_GMAXP   = OFF_QMAXP + B*CH1;         // B*CH1*C2
constexpr size_t OFF_SPMAXP  = OFF_GMAXP + (size_t)B*CH1*C2; // B*CH2
constexpr size_t OFF_XMP     = OFF_SPMAXP + B*CH2;        // B*CH2*C
constexpr size_t OFF_QMAX    = OFF_XMP + (size_t)B*CH2*C; // B
constexpr size_t OFF_INVQSUM = OFF_QMAX + B;              // B
constexpr size_t OFF_SPMAX   = OFF_INVQSUM + B;           // B
constexpr size_t OFF_INVSPSUM= OFF_SPMAX + B;             // B
constexpr size_t OFF_VEFF    = OFF_INVSPSUM + B;          // B*C
constexpr size_t OFF_MVEC    = OFF_VEFF + (size_t)B*C;    // B*C
constexpr size_t OFF_SVEC    = OFF_MVEC + (size_t)B*C;    // B*C
constexpr size_t OFF_TPART   = OFF_SVEC + (size_t)B*C;    // B*NK3*C*C
constexpr size_t OFF_ZPART   = OFF_TPART + (size_t)B*NK3*C*C; // B*NK3*C
// total ~3.29M floats (~13.1 MB)

__device__ __forceinline__ float wave_max(float v) {
#pragma unroll
  for (int o = 16; o; o >>= 1) v = fmaxf(v, __shfl_xor(v, o, 32));
  return v;
}
__device__ __forceinline__ float wave_sum(float v) {
#pragma unroll
  for (int o = 16; o; o >>= 1) v += __shfl_xor(v, o, 32);
  return v;
}

// ---------------------------------------------------------------------------
// Pass 1: q[b,p] = wq . x[:,p]  (store);  gmax partials of (Wql x)
// ---------------------------------------------------------------------------
__global__ void __launch_bounds__(256) k_pass1(const float* __restrict__ x,
                                               const float* __restrict__ wq,
                                               const float* __restrict__ wql,
                                               float* __restrict__ ws) {
  __shared__ float s_wq[C];
  __shared__ float s_wqlT[C * C2];   // transposed: [c][k] for vectorizable LDS reads
  __shared__ float s_red[8 * C2];
  __shared__ float s_red1[8];
  const int tid = threadIdx.x, lane = tid & 31, wid = tid >> 5;
  const int b = blockIdx.x / CH1, ch = blockIdx.x % CH1;

  for (int i = tid; i < C; i += 256) s_wq[i] = wq[i];
  for (int i = tid; i < C * C2; i += 256) {
    int c = i >> 5, k = i & 31;
    s_wqlT[i] = wql[k * C + c];
  }
  __syncthreads();

  const int p = ch * PX1 + tid * 2;
  const float* xb = x + (size_t)b * C * HW;
  float2 qv = make_float2(0.f, 0.f);
  float2 g[C2];
#pragma unroll
  for (int k = 0; k < C2; k++) g[k] = make_float2(0.f, 0.f);

  for (int c = 0; c < C; c++) {
    float2 xv = *(const float2*)(xb + (size_t)c * HW + p);
    qv.x += s_wq[c] * xv.x;
    qv.y += s_wq[c] * xv.y;
#pragma unroll
    for (int k = 0; k < C2; k++) {
      float wv = s_wqlT[c * C2 + k];
      g[k].x += wv * xv.x;
      g[k].y += wv * xv.y;
    }
  }
  *(float2*)(ws + OFF_Q + (size_t)b * HW + p) = qv;

  float qm = wave_max(fmaxf(qv.x, qv.y));
  if (lane == 0) s_red1[wid] = qm;
#pragma unroll
  for (int k = 0; k < C2; k++) {
    float gm = wave_max(fmaxf(g[k].x, g[k].y));
    if (lane == 0) s_red[wid * C2 + k] = gm;
  }
  __syncthreads();
  if (tid == 0) {
    float m = s_red1[0];
    for (int w = 1; w < 8; w++) m = fmaxf(m, s_red1[w]);
    ws[OFF_QMAXP + b * CH1 + ch] = m;
  }
  if (tid < C2) {
    float m = -1e30f;
    for (int w = 0; w < 8; w++) m = fmaxf(m, s_red[w * C2 + tid]);
    ws[OFF_GMAXP + (size_t)(b * CH1 + ch) * C2 + tid] = m;
  }
}

// ---------------------------------------------------------------------------
// Combine 1: qmax, 1/qsum, avg = softmax(gmax), v_eff = avg . Wvl
// ---------------------------------------------------------------------------
__global__ void __launch_bounds__(256) k_combine1(const float* __restrict__ wvl,
                                                  float* __restrict__ ws) {
  __shared__ float s_qmax;
  __shared__ float s_avg[C2];
  __shared__ float s_tmp[8];
  const int tid = threadIdx.x, lane = tid & 31, wid = tid >> 5;
  const int b = blockIdx.x;

  if (tid < 32) {   // CH1 == 32: wave 0 reduces qmax partials
    float v = ws[OFF_QMAXP + b * CH1 + tid];
    v = wave_max(v);
    if (tid == 0) s_qmax = v;
  }
  if (tid < C2) {   // gmax over chunks, then softmax over k (wave 0)
    float gm = -1e30f;
    for (int chk = 0; chk < CH1; chk++)
      gm = fmaxf(gm, ws[OFF_GMAXP + (size_t)(b * CH1 + chk) * C2 + tid]);
    float mx = wave_max(gm);
    float e = __expf(gm - mx);
    float ssum = wave_sum(e);
    s_avg[tid] = e / ssum;
  }
  __syncthreads();

  const float qmax = s_qmax;
  const float* qb = ws + OFF_Q + (size_t)b * HW;
  float acc = 0.f;
  for (int p = tid; p < HW; p += 256) acc += __expf(qb[p] - qmax);
  acc = wave_sum(acc);
  if (lane == 0) s_tmp[wid] = acc;
  __syncthreads();
  if (tid == 0) {
    float t = 0.f;
    for (int w = 0; w < 8; w++) t += s_tmp[w];
    ws[OFF_QMAX + b] = qmax;
    ws[OFF_INVQSUM + b] = 1.f / t;
  }
  if (tid < C) {
    float a = 0.f;
    for (int k = 0; k < C2; k++) a += s_avg[k] * wvl[k * C + tid];
    ws[OFF_VEFF + (size_t)b * C + tid] = a;
  }
}

// ---------------------------------------------------------------------------
// Pass 2: xm[c] partials (= sum_p x*mask), sp[p] = veff.x (store), spmax partials
// ---------------------------------------------------------------------------
__global__ void __launch_bounds__(256) k_pass2(const float* __restrict__ x,
                                               float* __restrict__ ws) {
  __shared__ float s_veff[C];
  __shared__ float s_xm[8 * C];
  __shared__ float s_red1[8];
  const int tid = threadIdx.x, lane = tid & 31, wid = tid >> 5;
  const int b = blockIdx.x / CH2, ch = blockIdx.x % CH2;

  if (tid < C) s_veff[tid] = ws[OFF_VEFF + (size_t)b * C + tid];
  const float qmax = ws[OFF_QMAX + b];
  const float invqsum = ws[OFF_INVQSUM + b];
  __syncthreads();

  const int p = ch * PX2 + tid * 2;
  const float* xb = x + (size_t)b * C * HW;
  float2 qv = *(const float2*)(ws + OFF_Q + (size_t)b * HW + p);
  const float m0 = __expf(qv.x - qmax) * invqsum;
  const float m1 = __expf(qv.y - qmax) * invqsum;

  float xm[C];
  float sp0 = 0.f, sp1 = 0.f;
#pragma unroll
  for (int c = 0; c < C; c++) {
    float2 xv = *(const float2*)(xb + (size_t)c * HW + p);
    xm[c] = xv.x * m0 + xv.y * m1;
    sp0 += s_veff[c] * xv.x;
    sp1 += s_veff[c] * xv.y;
  }
  *(float2*)(ws + OFF_SP + (size_t)b * HW + p) = make_float2(sp0, sp1);

  float sm = wave_max(fmaxf(sp0, sp1));
  if (lane == 0) s_red1[wid] = sm;
#pragma unroll
  for (int c = 0; c < C; c++) {
    float r = wave_sum(xm[c]);
    if (lane == 0) s_xm[wid * C + c] = r;
  }
  __syncthreads();
  if (tid == 0) {
    float m = s_red1[0];
    for (int w = 1; w < 8; w++) m = fmaxf(m, s_red1[w]);
    ws[OFF_SPMAXP + b * CH2 + ch] = m;
  }
  if (tid < C) {
    float a = 0.f;
    for (int w = 0; w < 8; w++) a += s_xm[w * C + tid];
    ws[OFF_XMP + (size_t)(b * CH2 + ch) * C + tid] = a;
  }
}

// ---------------------------------------------------------------------------
// Combine 2: spmax, 1/spsum, then the tiny MLP chain -> m[b,c]
// ---------------------------------------------------------------------------
__global__ void __launch_bounds__(256) k_combine2(
    const float* __restrict__ wvr, const float* __restrict__ lng,
    const float* __restrict__ lnb, const float* __restrict__ wup,
    const float* __restrict__ wsk1, const float* __restrict__ bng,
    const float* __restrict__ bnb, const float* __restrict__ wsk2,
    float* __restrict__ ws) {
  __shared__ float s_spmax;
  __shared__ float s_xm[C], s_ctx[C2], s_ca[C], s_u[C], s_z[C2], s_z2[C];
  __shared__ float s_tmp[8], s_sum2[2];
  const int tid = threadIdx.x, lane = tid & 31, wid = tid >> 5;
  const int b = blockIdx.x;

  if (tid < 32) {   // CH2 == 32
    float v = ws[OFF_SPMAXP + b * CH2 + tid];
    v = wave_max(v);
    if (tid == 0) s_spmax = v;
  }
  if (tid < C) {
    float a = 0.f;
    for (int chk = 0; chk < CH2; chk++)
      a += ws[OFF_XMP + (size_t)(b * CH2 + chk) * C + tid];
    s_xm[tid] = a;
  }
  __syncthreads();

  const float spmax = s_spmax;
  const float* spb = ws + OFF_SP + (size_t)b * HW;
  float acc = 0.f;
  for (int p = tid; p < HW; p += 256) acc += __expf(spb[p] - spmax);
  acc = wave_sum(acc);
  if (lane == 0) s_tmp[wid] = acc;
  __syncthreads();
  if (tid == 0) {
    float t = 0.f;
    for (int w = 0; w < 8; w++) t += s_tmp[w];
    ws[OFF_SPMAX + b] = spmax;
    ws[OFF_INVSPSUM + b] = 1.f / t;
  }

  // context = Wvr . xm ; LayerNorm(32) ; relu   (wave 0)
  if (tid < C2) {
    float c0 = 0.f;
    for (int c = 0; c < C; c++) c0 += wvr[tid * C + c] * s_xm[c];
    float mu = wave_sum(c0) * (1.f / C2);
    float d = c0 - mu;
    float var = wave_sum(d * d) * (1.f / C2);
    float ctx = d * rsqrtf(var + 1e-5f) * lng[tid] + lnb[tid];
    s_ctx[tid] = fmaxf(ctx, 0.f);
  }
  __syncthreads();
  if (tid < C) {
    float z = 0.f;
    for (int k = 0; k < C2; k++) z += wup[tid * C2 + k] * s_ctx[k];
    float ca = 1.f / (1.f + __expf(-z));
    s_ca[tid] = ca;
    s_u[tid] = ca * (1.f + 1.f / HW) + 1.f / HW;   // mean(seq+par) closed form
  }
  __syncthreads();
  if (tid < C2) {
    float z = 0.f;
    for (int c = 0; c < C; c++) z += wsk1[tid * C + c] * s_u[c];
    z = z * rsqrtf(1.f + 1e-5f) * bng[tid] + bnb[tid];
    s_z[tid] = fmaxf(z, 0.f);
  }
  __syncthreads();
  if (tid < C) {
    float z = 0.f;
    for (int k = 0; k < C2; k++) z += wsk2[tid * C2 + k] * s_z[k];
    s_z2[tid] = z;
    float mx = wave_max(z);
    if (lane == 0) s_tmp[wid] = mx;
  }
  __syncthreads();
  if (tid < C) {
    float mx = fmaxf(s_tmp[0], s_tmp[1]);
    float e = __expf(s_z2[tid] - mx);
    float ssum = wave_sum(e);
    if (lane == 0) s_sum2[wid] = ssum;
  }
  __syncthreads();
  if (tid < C) {
    float mx = fmaxf(s_tmp[0], s_tmp[1]);
    float a = __expf(s_z2[tid] - mx) / (s_sum2[0] + s_sum2[1]);
    // sq_pr logits = m*sa + const ; const cancels in pixel softmax
    ws[OFF_MVEC + (size_t)b * C + tid] = a * s_ca[tid] + 1.f - a;
  }
}

// ---------------------------------------------------------------------------
// Pass 3 (WMMA): T[b] += E(64 x K) * X^T(K x 64), Z[b] += rowsum(E)
//   E[c,p] = exp(m[c] * (sa[p]-samax)); staged t = sa-samax in LDS.
// A layout (ISA 7.12.2, 16-bit A 16x32): lane<16 -> K 0-7 & 16-23;
// lane>=16 -> K 8-15 & 24-31. B assumed symmetric with lane = column.
// ---------------------------------------------------------------------------
__global__ void __launch_bounds__(128) k_pass3(const float* __restrict__ x,
                                               float* __restrict__ ws) {
  __shared__ float s_t[PX3];
  const int tid = threadIdx.x;
  const int b = blockIdx.x / NK3, ck = blockIdx.x % NK3;
  const int pbase = ck * PX3;

  const float spmax = ws[OFF_SPMAX + b];
  const float invspsum = ws[OFF_INVSPSUM + b];
  const float* spb = ws + OFF_SP + (size_t)b * HW + pbase;
  for (int i = tid; i < PX3; i += 128)
    s_t[i] = (__expf(spb[i] - spmax) - 1.f) * invspsum;  // sa[p] - 1/spsum
  __syncthreads();

  const int lane = tid & 31;
  const int wv = tid >> 5;          // wave id 0..3 -> M-tile rows [16w,16w+16)
  const int lr = lane & 15;
  const int kb = (lane >> 4) * 8;   // K sub-block base within 32
  const int row = wv * 16 + lr;
  const float mrow = ws[OFF_MVEC + (size_t)b * C + row];
  const float* xb = x + (size_t)b * C * HW + pbase;

  v8f acc[4] = {};
  float zacc = 0.f;

  for (int p0 = 0; p0 < PX3; p0 += 32) {
    // --- A tile: E values for this lane's row ---
    v16h av;
#pragma unroll
    for (int i = 0; i < 8; i++) {
      float e0 = __expf(mrow * s_t[p0 + kb + i]);
      float e1 = __expf(mrow * s_t[p0 + kb + 16 + i]);
      zacc += e0 + e1;
      av[i]     = (_Float16)e0;
      av[8 + i] = (_Float16)e1;
    }
    // --- B tiles: columns of X^T = rows of x (contiguous pixels) ---
#pragma unroll
    for (int j = 0; j < 4; j++) {
      const float* xc = xb + (size_t)(j * 16 + lr) * HW + p0 + kb;
      float4 b0 = *(const float4*)(xc);
      float4 b1 = *(const float4*)(xc + 4);
      float4 b2 = *(const float4*)(xc + 16);
      float4 b3 = *(const float4*)(xc + 20);
      v16h bv;
      bv[0]  = (_Float16)b0.x; bv[1]  = (_Float16)b0.y;
      bv[2]  = (_Float16)b0.z; bv[3]  = (_Float16)b0.w;
      bv[4]  = (_Float16)b1.x; bv[5]  = (_Float16)b1.y;
      bv[6]  = (_Float16)b1.z; bv[7]  = (_Float16)b1.w;
      bv[8]  = (_Float16)b2.x; bv[9]  = (_Float16)b2.y;
      bv[10] = (_Float16)b2.z; bv[11] = (_Float16)b2.w;
      bv[12] = (_Float16)b3.x; bv[13] = (_Float16)b3.y;
      bv[14] = (_Float16)b3.z; bv[15] = (_Float16)b3.w;
      acc[j] = __builtin_amdgcn_wmma_f32_16x16x32_f16(
          false, av, false, bv, (short)0, acc[j], false, false);
    }
  }

  // store partial T (deterministic, no atomics): D layout -> VGPR r,
  // lanes 0-15: (M=r, N=lane); lanes 16-31: (M=r+8, N=lane-16)
  float* Tp = ws + OFF_TPART + (size_t)(b * NK3 + ck) * C * C;
#pragma unroll
  for (int j = 0; j < 4; j++) {
#pragma unroll
    for (int r = 0; r < 8; r++) {
      int rr = wv * 16 + r + 8 * (lane >> 4);
      Tp[(size_t)rr * C + j * 16 + lr] = acc[j][r];
    }
  }
  float zo = zacc + __shfl_xor(zacc, 16, 32);
  if (lane < 16) ws[OFF_ZPART + (size_t)(b * NK3 + ck) * C + row] = zo;
}

// ---------------------------------------------------------------------------
// Finalize: s[b,o] = (sum_{c'} Wres[o,c'] * T[b,o,c']) / Z[b,o]
// ---------------------------------------------------------------------------
__global__ void __launch_bounds__(64) k_fin_s(const float* __restrict__ wres,
                                              float* __restrict__ ws) {
  const int b = blockIdx.x, o = threadIdx.x;
  float z = 0.f;
  for (int ck = 0; ck < NK3; ck++)
    z += ws[OFF_ZPART + (size_t)(b * NK3 + ck) * C + o];
  float s = 0.f;
  for (int c = 0; c < C; c++) {
    float tv = 0.f;
    for (int ck = 0; ck < NK3; ck++)
      tv += ws[OFF_TPART + (size_t)(b * NK3 + ck) * C * C + (size_t)o * C + c];
    s += wres[o * C + c] * tv;
  }
  ws[OFF_SVEC + (size_t)b * C + o] = s / z;
}

// ---------------------------------------------------------------------------
// out = x + s (broadcast over pixels), float4-vectorized
// ---------------------------------------------------------------------------
__global__ void __launch_bounds__(256) k_add(const float* __restrict__ x,
                                             const float* __restrict__ ws,
                                             float* __restrict__ out, int n4) {
  int i = blockIdx.x * 256 + threadIdx.x;
  if (i >= n4) return;
  size_t e = (size_t)i * 4;
  int bc = (int)(e / HW);   // flat (b*C + c)
  float s = ws[OFF_SVEC + bc];
  float4 xv = *(const float4*)(x + e);
  float4 o = make_float4(xv.x + s, xv.y + s, xv.z + s, xv.w + s);
  *(float4*)(out + e) = o;
}

// ---------------------------------------------------------------------------
extern "C" void kernel_launch(void* const* d_in, const int* in_sizes, int n_in,
                              void* d_out, int out_size, void* d_ws, size_t ws_size,
                              hipStream_t stream) {
  const float* x    = (const float*)d_in[0];
  const float* wqr  = (const float*)d_in[1];   // w_q_right (1,64)
  const float* wvr  = (const float*)d_in[2];   // w_v_right (32,64)
  const float* lng  = (const float*)d_in[3];
  const float* lnb  = (const float*)d_in[4];
  const float* wup  = (const float*)d_in[5];   // (64,32)
  const float* wql  = (const float*)d_in[6];   // w_q_left (32,64)
  const float* wvl  = (const float*)d_in[7];   // w_v_left (32,64)
  const float* wsk1 = (const float*)d_in[8];   // (32,64)
  const float* bng  = (const float*)d_in[9];
  const float* bnb  = (const float*)d_in[10];
  const float* wsk2 = (const float*)d_in[11];  // (64,32)
  const float* wres = (const float*)d_in[12];  // (64,64)
  float* out = (float*)d_out;
  float* ws  = (float*)d_ws;   // needs ~13.2 MB; every region is write-before-read

  k_pass1<<<B * CH1, 256, 0, stream>>>(x, wqr, wql, ws);
  k_combine1<<<B, 256, 0, stream>>>(wvl, ws);
  k_pass2<<<B * CH2, 256, 0, stream>>>(x, ws);
  k_combine2<<<B, 256, 0, stream>>>(wvr, lng, lnb, wup, wsk1, bng, bnb, wsk2, ws);
  k_pass3<<<B * NK3, 128, 0, stream>>>(x, ws);
  k_fin_s<<<B, 64, 0, stream>>>(wres, ws);
  int n4 = B * C * HW / 4;
  k_add<<<(n4 + 255) / 256, 256, 0, stream>>>(x, ws, out, n4);
}
